// GPT_34359739051
// MI455X (gfx1250) — matrix operand order repaired and checked by reading, not compiled, not run on hardware
//
#include <hip/hip_runtime.h>
#include <math.h>

// ---------------------------------------------------------------------------
// GPT forward on MI455X (gfx1250, wave32). All GEMMs via v_wmma_f32_16x16x32_f16
// (f16 inputs, fp32 accumulate). Weights pre-transposed+converted to f16 [N][K].
// GEMM K-stages of 64 are staged into double-buffered LDS by the Tensor Data
// Mover (tensor_load_to_lds, pipelined against WMMA with s_wait_tensorcnt<=2),
// falling back to b128 copies if the builtin is missing. Head-GEMM logits are
// streamed with nontemporal stores so the 824MB output doesn't evict the 51MB
// head-weight tile set from the 192MB L2. Residual/softmax/LN/loss in fp32.
// ---------------------------------------------------------------------------

typedef _Float16 half8  __attribute__((ext_vector_type(8)));
typedef _Float16 half16 __attribute__((ext_vector_type(16)));
typedef float    float8 __attribute__((ext_vector_type(8)));

#define ACT_NONE 0
#define ACT_GELU 1

#if defined(__AMDGCN__) && __has_builtin(__builtin_amdgcn_tensor_load_to_lds)
#define HAS_TDM 1
#else
#define HAS_TDM 0
#endif

#if HAS_TDM
typedef unsigned int u32x4 __attribute__((ext_vector_type(4)));
typedef int          i32x4 __attribute__((ext_vector_type(4)));
typedef int          i32x8 __attribute__((ext_vector_type(8)));

// 2D f16 tile DMA: global (row-major, rowStride elems) -> LDS (contiguous
// tileW*tileH). Rows/cols past (dim0, dim1) read as zero (hardware OOB fill).
__device__ __forceinline__ void tdm_load_tile_f16(void* ldsDst, const _Float16* gsrc,
                                                  unsigned tileW, unsigned tileH,
                                                  unsigned dim0, unsigned dim1,
                                                  unsigned rowStrideElems) {
  unsigned long long ga = (unsigned long long)(uintptr_t)gsrc;
  unsigned lds = (unsigned)(uintptr_t)ldsDst;     // low 32 bits = LDS byte offset
  u32x4 g0;
  g0[0] = 1u;                                      // count=1 (valid user D#)
  g0[1] = lds;                                     // lds_addr
  g0[2] = (unsigned)(ga & 0xFFFFFFFFu);            // global_addr[31:0]
  g0[3] = (unsigned)((ga >> 32) & 0x01FFFFFFu) | 0x80000000u; // addr[56:32], type=2
  unsigned long long st = (unsigned long long)rowStrideElems;
  i32x8 g1;
  g1[0] = 0x00010000;                              // data_size=1 -> 2 bytes/elem
  g1[1] = (int)((dim0 & 0xFFFFu) << 16);           // tensor_dim0[15:0] @ bits63:48
  g1[2] = (int)((dim0 >> 16) | ((dim1 & 0xFFFFu) << 16));   // dim0 hi, dim1 lo
  g1[3] = (int)(((dim1 >> 16) & 0xFFFFu) | (tileW << 16));  // dim1 hi, tile_dim0
  g1[4] = (int)(tileH & 0xFFFFu);                  // tile_dim1 (tile_dim2 = 0)
  g1[5] = (int)(st & 0xFFFFFFFFu);                 // tensor_dim0_stride[31:0]
  g1[6] = (int)((st >> 32) & 0xFFFFu);             // dim0_stride[47:32]
  g1[7] = 0;
  i32x4 z4 = {};
#if __clang_major__ >= 23
  i32x8 z8 = {};
  __builtin_amdgcn_tensor_load_to_lds(g0, g1, z4, z4, z8, 0);
#else
  __builtin_amdgcn_tensor_load_to_lds(g0, g1, z4, z4, 0);
#endif
}
#endif  // HAS_TDM

// ------------------------ weight transpose/convert ------------------------
// Wt[n*K + k] = (f16) W[k*N + n];  K is a power of two (kshift).
__global__ __launch_bounds__(256)
void transpose_f16_kernel(const float* __restrict__ W, _Float16* __restrict__ Wt,
                          int N, int kshift, long total) {
  long i = (long)blockIdx.x * 256 + threadIdx.x;
  if (i >= total) return;
  int K = 1 << kshift;
  int n = (int)(i >> kshift);
  int k = (int)(i & (K - 1));
  Wt[i] = (_Float16)W[(size_t)k * N + n];
}

// ------------------------------ embeddings --------------------------------
__global__ void embed_kernel(const int* __restrict__ idx,
                             const float* __restrict__ tok,
                             const float* __restrict__ pos,
                             float* __restrict__ x, int T, int C) {
  int row = blockIdx.x;               // b*T + t
  int t = row % T;
  int token = idx[row];
  const float* te = tok + (size_t)token * C;
  const float* pe = pos + (size_t)t * C;
  float* xr = x + (size_t)row * C;
  for (int c = threadIdx.x; c < C; c += blockDim.x) xr[c] = te[c] + pe[c];
}

// --------------------------- layernorm (f16 out) --------------------------
__global__ __launch_bounds__(256)
void ln_kernel(const float* __restrict__ x, const float* __restrict__ w,
               const float* __restrict__ b, _Float16* __restrict__ out, int C) {
  int row = blockIdx.x;
  const float* xr = x + (size_t)row * C;
  _Float16* outr = out + (size_t)row * C;
  __shared__ float red[256];
  int tid = threadIdx.x;

  float s = 0.f;
  for (int c = tid; c < C; c += 256) s += xr[c];
  red[tid] = s; __syncthreads();
  for (int off = 128; off > 0; off >>= 1) {
    if (tid < off) red[tid] += red[tid + off];
    __syncthreads();
  }
  float mean = red[0] / (float)C;
  __syncthreads();

  float v = 0.f;
  for (int c = tid; c < C; c += 256) { float d = xr[c] - mean; v += d * d; }
  red[tid] = v; __syncthreads();
  for (int off = 128; off > 0; off >>= 1) {
    if (tid < off) red[tid] += red[tid + off];
    __syncthreads();
  }
  float inv = rsqrtf(red[0] / (float)C + 1e-5f);

  for (int c = tid; c < C; c += 256)
    outr[c] = (_Float16)((xr[c] - mean) * inv * w[c] + b[c]);
}

// ------------------------------ WMMA GEMM ---------------------------------
// out = act(A[M,K] @ Wt^T + bias) (+resid). A f16 [M][K]; Wt f16 [N][K].
// Block tile 64x64, 8 waves; K-stages of 64 double-buffered in LDS, DMA'd by
// TDM and pipelined one stage ahead of the WMMAs.
__global__ __launch_bounds__(256)
void gemm_wmma(const _Float16* __restrict__ A, const _Float16* __restrict__ Wt,
               const float* __restrict__ bias, const float* __restrict__ resid,
               float* __restrict__ outF, _Float16* __restrict__ outH,
               int M, int N, int K, int act, int ntOut) {
  __shared__ alignas(16) _Float16 As[2][64 * 64];   // [buf][row][k]
  __shared__ alignas(16) _Float16 Ws[2][64 * 64];   // [buf][n][k]

  int tid  = threadIdx.x;
  int lane = tid & 31;
  int wave = tid >> 5;
  int wr = wave & 3;            // 4 row tiles of 16
  int wc = wave >> 2;           // 2 col groups of 32
  int bm = blockIdx.y * 64;
  int bn = blockIdx.x * 64;

  int nlo = lane & 15;
  int hh  = lane >> 4;
  int koffA = hh * 8;
  int koffB = hh * 8;

  float8 acc0 = {}; float8 acc1 = {};

  int nst = K >> 6;             // stages of 64

#if HAS_TDM
  if (wave == 0) {              // prologue: stage 0 into buffer 0
    tdm_load_tile_f16(&As[0][0], A + (size_t)bm * K, 64u, 64u,
                      (unsigned)K, (unsigned)(M - bm), (unsigned)K);
    tdm_load_tile_f16(&Ws[0][0], Wt + (size_t)bn * K, 64u, 64u,
                      (unsigned)K, (unsigned)(N - bn), (unsigned)K);
  }
#endif

  for (int s = 0; s < nst; ++s) {
    int cur = s & 1;
#if HAS_TDM
    if (wave == 0) {
      if (s + 1 < nst) {        // issue next stage, then wait for current
        int k0 = (s + 1) << 6;
        tdm_load_tile_f16(&As[cur ^ 1][0], A + (size_t)bm * K + k0, 64u, 64u,
                          (unsigned)(K - k0), (unsigned)(M - bm), (unsigned)K);
        tdm_load_tile_f16(&Ws[cur ^ 1][0], Wt + (size_t)bn * K + k0, 64u, 64u,
                          (unsigned)(K - k0), (unsigned)(N - bn), (unsigned)K);
        __builtin_amdgcn_s_wait_tensorcnt(2);   // in-order: stage s has landed
      } else {
        __builtin_amdgcn_s_wait_tensorcnt(0);
      }
    }
    __syncthreads();
#else
    {
      int k0 = s << 6;
      int r = tid >> 2, kk = (tid & 3) * 16;   // 64 rows x 4 chunks of 16 f16
      const _Float16* ag = A + (size_t)(bm + r) * K + k0 + kk;
      *(half8*)&As[cur][r * 64 + kk]     = *(const half8*)ag;
      *(half8*)&As[cur][r * 64 + kk + 8] = *(const half8*)(ag + 8);
      int gn = bn + r;
      half8 w0 = {}, w1 = {};
      if (gn < N) {
        const _Float16* wg = Wt + (size_t)gn * K + k0 + kk;
        w0 = *(const half8*)wg;
        w1 = *(const half8*)(wg + 8);
      }
      *(half8*)&Ws[cur][r * 64 + kk]     = w0;
      *(half8*)&Ws[cur][r * 64 + kk + 8] = w1;
    }
    __syncthreads();
#endif

    for (int kh = 0; kh < 2; ++kh) {
      // A fragment (rows wr*16 .. +15), k-half kh
      const _Float16* ap = &As[cur][(wr * 16 + nlo) * 64 + kh * 32 + koffA];
      half8 alo = *(const half8*)ap;
      half8 ahi = *(const half8*)(ap + 16);
      half16 af;
      for (int e = 0; e < 8; ++e) { af[e] = alo[e]; af[e + 8] = ahi[e]; }

      {
        const _Float16* bp = &Ws[cur][(wc * 32 + nlo) * 64 + kh * 32 + koffB];
        half8 blo = *(const half8*)bp;
        half8 bhi = *(const half8*)(bp + 16);
        half16 bf;
        for (int e = 0; e < 8; ++e) { bf[e] = blo[e]; bf[e + 8] = bhi[e]; }
        acc0 = __builtin_amdgcn_wmma_f32_16x16x32_f16(false, af, false, bf,
                                                      (short)0, acc0, false, false);
      }
      {
        const _Float16* bp = &Ws[cur][(wc * 32 + 16 + nlo) * 64 + kh * 32 + koffB];
        half8 blo = *(const half8*)bp;
        half8 bhi = *(const half8*)(bp + 16);
        half16 bf;
        for (int e = 0; e < 8; ++e) { bf[e] = blo[e]; bf[e + 8] = bhi[e]; }
        acc1 = __builtin_amdgcn_wmma_f32_16x16x32_f16(false, af, false, bf,
                                                      (short)0, acc1, false, false);
      }
    }
    __syncthreads();            // LDS reads done before next stage's DMA lands
  }

  // epilogue (C layout: n = lane&15, m = v + hh*8)
  for (int ct = 0; ct < 2; ++ct) {
    float8 acc = ct ? acc1 : acc0;
    int gn = bn + wc * 32 + ct * 16 + nlo;
    if (gn >= N) continue;
    float bv = bias ? bias[gn] : 0.f;
    for (int v = 0; v < 8; ++v) {
      int gm = bm + wr * 16 + hh * 8 + v;
      float val = acc[v] + bv;
      if (act == ACT_GELU) val = 0.5f * val * (1.f + erff(val * 0.70710678118f));
      if (resid) val += resid[(size_t)gm * N + gn];
      if (outF) {
        float* p = outF + (size_t)gm * N + gn;
        if (ntOut) __builtin_nontemporal_store(val, p);
        else *p = val;
      }
      if (outH) outH[(size_t)gm * N + gn] = (_Float16)val;
    }
  }
}

// --------------------------- flash attention ------------------------------
// One wave per (b, h, 16-row query tile). q/k/v/y are f16 [M][C].
__global__ __launch_bounds__(256)
void attn_kernel(const _Float16* __restrict__ q, const _Float16* __restrict__ k,
                 const _Float16* __restrict__ v, _Float16* __restrict__ y,
                 int T, int H, int Dh) {
  __shared__ alignas(16) _Float16 Pbuf[8][16 * 32];

  int lane = threadIdx.x & 31;
  int wave = threadIdx.x >> 5;
  int gw = blockIdx.x * 8 + wave;
  int qtiles = T >> 4;
  int qt = gw % qtiles;
  int hd = (gw / qtiles) % H;
  int b  = gw / (qtiles * H);
  int C  = H * Dh;

  int nlo = lane & 15;
  int hh  = lane >> 4;
  int koffA = hh * 8;
  int koffB = hh * 8;
  const float scale = rsqrtf((float)Dh);

  // Q fragments (A layout), D=64 -> two K=32 fragments
  const _Float16* qbase = q + (size_t)(b * T + qt * 16 + nlo) * C + hd * Dh;
  half16 qf0, qf1;
  {
    half8 a0 = *(const half8*)(qbase + koffA);
    half8 a1 = *(const half8*)(qbase + koffA + 16);
    half8 a2 = *(const half8*)(qbase + 32 + koffA);
    half8 a3 = *(const half8*)(qbase + 32 + koffA + 16);
    for (int e = 0; e < 8; ++e) {
      qf0[e] = a0[e]; qf0[e + 8] = a1[e];
      qf1[e] = a2[e]; qf1[e + 8] = a3[e];
    }
  }

  float8 acc[4];
  for (int c = 0; c < 4; ++c) acc[c] = (float8){};
  float mrow[8], lrow[8];
  for (int r = 0; r < 8; ++r) { mrow[r] = -1e30f; lrow[r] = 0.f; }

  _Float16* pb = &Pbuf[wave][0];

  int pmax = qt >> 1;
  for (int p = 0; p <= pmax; ++p) {
    float sv[2][8];
    for (int jj = 0; jj < 2; ++jj) {
      int jt = 2 * p + jj;
      const _Float16* kbase = k + (size_t)(b * T + jt * 16 + nlo) * C + hd * Dh;
      half16 kf0, kf1;
      {
        half8 b0 = *(const half8*)(kbase + koffB);
        half8 b1 = *(const half8*)(kbase + koffB + 16);
        half8 b2 = *(const half8*)(kbase + 32 + koffB);
        half8 b3 = *(const half8*)(kbase + 32 + koffB + 16);
        for (int e = 0; e < 8; ++e) {
          kf0[e] = b0[e]; kf0[e + 8] = b1[e];
          kf1[e] = b2[e]; kf1[e + 8] = b3[e];
        }
      }
      float8 s = {};
      s = __builtin_amdgcn_wmma_f32_16x16x32_f16(false, qf0, false, kf0,
                                                 (short)0, s, false, false);
      s = __builtin_amdgcn_wmma_f32_16x16x32_f16(false, qf1, false, kf1,
                                                 (short)0, s, false, false);
      int kcol = jt * 16 + nlo;
      for (int r = 0; r < 8; ++r) {
        int qrow = qt * 16 + hh * 8 + r;
        float xs = s[r] * scale;
        sv[jj][r] = (kcol <= qrow) ? xs : -1e30f;
      }
    }
    // online softmax (row = hh*8+r; 16 keys/tile live across a half-wave)
    float fac[8];
    for (int r = 0; r < 8; ++r) {
      float tm = fmaxf(sv[0][r], sv[1][r]);
      for (int m = 8; m >= 1; m >>= 1) tm = fmaxf(tm, __shfl_xor(tm, m, 32));
      float mnew = fmaxf(mrow[r], tm);
      float p0 = __expf(sv[0][r] - mnew);
      float p1 = __expf(sv[1][r] - mnew);
      sv[0][r] = p0; sv[1][r] = p1;
      float ls = p0 + p1;
      for (int m = 8; m >= 1; m >>= 1) ls += __shfl_xor(ls, m, 32);
      float f = __expf(mrow[r] - mnew);
      lrow[r] = lrow[r] * f + ls;
      mrow[r] = mnew;
      fac[r] = f;
    }
    for (int c = 0; c < 4; ++c)
      for (int r = 0; r < 8; ++r) acc[c][r] *= fac[r];

    // P (16x32) C-layout -> LDS -> A-layout
    for (int jj = 0; jj < 2; ++jj)
      for (int r = 0; r < 8; ++r)
        pb[(hh * 8 + r) * 32 + jj * 16 + nlo] = (_Float16)sv[jj][r];
    asm volatile("s_wait_dscnt 0" ::: "memory");

    const _Float16* pr = pb + nlo * 32 + koffA;
    half8 plo = *(const half8*)pr;
    half8 phi = *(const half8*)(pr + 16);
    half16 pf;
    for (int e = 0; e < 8; ++e) { pf[e] = plo[e]; pf[e + 8] = phi[e]; }

    // acc += P @ V over 32 keys, 4 column tiles of 16
    for (int c = 0; c < 4; ++c) {
      const _Float16* vbase = v + (size_t)(b * T + 32 * p + koffB) * C
                                + hd * Dh + c * 16 + nlo;
      half16 vf;
      for (int e = 0; e < 16; ++e) vf[e] = vbase[(size_t)(e < 8 ? e : e + 8) * C];
      acc[c] = __builtin_amdgcn_wmma_f32_16x16x32_f16(false, pf, false, vf,
                                                      (short)0, acc[c], false, false);
    }
  }

  for (int c = 0; c < 4; ++c)
    for (int r = 0; r < 8; ++r) {
      int gm = b * T + qt * 16 + hh * 8 + r;
      y[(size_t)gm * C + hd * Dh + c * 16 + nlo] = (_Float16)(acc[c][r] / lrow[r]);
    }
}

// ------------------------------ loss --------------------------------------
__global__ void zero_loss(float* loss) { *loss = 0.f; }

__global__ __launch_bounds__(256)
void loss_kernel(const float* __restrict__ logits, const int* __restrict__ targets,
                 float* __restrict__ loss, int V, float invMT) {
  int row = blockIdx.x;
  const float* lr = logits + (size_t)row * V;
  __shared__ float red[256];
  int tid = threadIdx.x;

  float mx = -1e30f;
  for (int c = tid; c < V; c += 256)
    mx = fmaxf(mx, __builtin_nontemporal_load(lr + c));
  red[tid] = mx; __syncthreads();
  for (int off = 128; off > 0; off >>= 1) {
    if (tid < off) red[tid] = fmaxf(red[tid], red[tid + off]);
    __syncthreads();
  }
  mx = red[0]; __syncthreads();

  float s = 0.f;
  for (int c = tid; c < V; c += 256)
    s += __expf(__builtin_nontemporal_load(lr + c) - mx);
  red[tid] = s; __syncthreads();
  for (int off = 128; off > 0; off >>= 1) {
    if (tid < off) red[tid] += red[tid + off];
    __syncthreads();
  }
  if (tid == 0) {
    float lse = mx + __logf(red[0]);
    float lp = lr[targets[row]] - lse;
    atomicAdd(loss, -lp * invMT);
  }
}

// ------------------------------ driver ------------------------------------
extern "C" void kernel_launch(void* const* d_in, const int* in_sizes, int n_in,
                              void* d_out, int out_size, void* d_ws, size_t ws_size,
                              hipStream_t stream) {
  (void)in_sizes; (void)n_in; (void)out_size; (void)ws_size;

  const int*   idx     = (const int*)d_in[0];
  const int*   targets = (const int*)d_in[1];
  const float* tok     = (const float*)d_in[2];
  const float* pos     = (const float*)d_in[3];
  const float* Wq  = (const float*)d_in[4];
  const float* bq  = (const float*)d_in[5];
  const float* Wk  = (const float*)d_in[6];
  const float* bk  = (const float*)d_in[7];
  const float* Wv  = (const float*)d_in[8];
  const float* bv  = (const float*)d_in[9];
  const float* Wo  = (const float*)d_in[10];
  const float* bo  = (const float*)d_in[11];
  const float* ln1w = (const float*)d_in[12];
  const float* ln1b = (const float*)d_in[13];
  const float* W1  = (const float*)d_in[14];
  const float* b1  = (const float*)d_in[15];
  const float* W2  = (const float*)d_in[16];
  const float* b2  = (const float*)d_in[17];
  const float* lnfw = (const float*)d_in[18];
  const float* lnfb = (const float*)d_in[19];
  const float* headW = (const float*)d_in[20];

  const int Bn = 2, T = 2048, C = 512, H = 8, Dh = 64, L = 4, F = 2048, V = 50257;
  const int M = Bn * T;               // 4096

  char* base = (char*)d_ws;
  float*     x    = (float*)base;     base += (size_t)M * C * 4;
  _Float16*  hbuf = (_Float16*)base;  base += (size_t)M * C * 2;
  _Float16*  qb   = (_Float16*)base;  base += (size_t)M * C * 2;
  _Float16*  kb   = (_Float16*)base;  base += (size_t)M * C * 2;
  _Float16*  vb   = (_Float16*)base;  base += (size_t)M * C * 2;
  _Float16*  yb   = (_Float16*)base;  base += (size_t)M * C * 2;
  _Float16*  hid  = (_Float16*)base;  base += (size_t)M * F * 2;
  _Float16*  WqT  = (_Float16*)base;  base += (size_t)L * C * C * 2;
  _Float16*  WkT  = (_Float16*)base;  base += (size_t)L * C * C * 2;
  _Float16*  WvT  = (_Float16*)base;  base += (size_t)L * C * C * 2;
  _Float16*  WoT  = (_Float16*)base;  base += (size_t)L * C * C * 2;
  _Float16*  W1T  = (_Float16*)base;  base += (size_t)L * C * F * 2;  // [F][C]
  _Float16*  W2T  = (_Float16*)base;  base += (size_t)L * F * C * 2;  // [C][F]
  _Float16*  hWT  = (_Float16*)base;  base += (size_t)V * C * 2;      // [V][C]

  float* logits = (float*)d_out;
  float* loss   = logits + (size_t)M * V;

  // ---- one-time weight transpose/convert to f16 [N][K] ----
  long tCC = (long)C * C;
  long tCF = (long)C * F;
  int  bCC = (int)((tCC + 255) / 256);
  int  bCF = (int)((tCF + 255) / 256);
  for (int l = 0; l < L; ++l) {
    transpose_f16_kernel<<<bCC, 256, 0, stream>>>(Wq + (size_t)l * tCC, WqT + (size_t)l * tCC, C, 9, tCC);
    transpose_f16_kernel<<<bCC, 256, 0, stream>>>(Wk + (size_t)l * tCC, WkT + (size_t)l * tCC, C, 9, tCC);
    transpose_f16_kernel<<<bCC, 256, 0, stream>>>(Wv + (size_t)l * tCC, WvT + (size_t)l * tCC, C, 9, tCC);
    transpose_f16_kernel<<<bCC, 256, 0, stream>>>(Wo + (size_t)l * tCC, WoT + (size_t)l * tCC, C, 9, tCC);
    transpose_f16_kernel<<<bCF, 256, 0, stream>>>(W1 + (size_t)l * tCF, W1T + (size_t)l * tCF, F, 9,  tCF); // [F][C], K=C
    transpose_f16_kernel<<<bCF, 256, 0, stream>>>(W2 + (size_t)l * tCF, W2T + (size_t)l * tCF, C, 11, tCF); // [C][F], K=F
  }
  long tHV = (long)V * C;
  transpose_f16_kernel<<<(int)((tHV + 255) / 256), 256, 0, stream>>>(headW, hWT, V, 9, tHV);

  // ---- forward pass ----
  embed_kernel<<<M, 256, 0, stream>>>(idx, tok, pos, x, T, C);

  dim3 gC(C / 64, M / 64);
  dim3 gF(F / 64, M / 64);
  int attnBlocks = (Bn * H * (T / 16)) / 8;     // 256

  for (int l = 0; l < L; ++l) {
    const float* w1l = ln1w + (size_t)l * C;
    const float* b1l = ln1b + (size_t)l * C;
    ln_kernel<<<M, 256, 0, stream>>>(x, w1l, b1l, hbuf, C);
    gemm_wmma<<<gC, 256, 0, stream>>>(hbuf, WqT + (size_t)l * tCC, bq + (size_t)l * C,
                                      nullptr, nullptr, qb, M, C, C, ACT_NONE, 0);
    gemm_wmma<<<gC, 256, 0, stream>>>(hbuf, WkT + (size_t)l * tCC, bk + (size_t)l * C,
                                      nullptr, nullptr, kb, M, C, C, ACT_NONE, 0);
    gemm_wmma<<<gC, 256, 0, stream>>>(hbuf, WvT + (size_t)l * tCC, bv + (size_t)l * C,
                                      nullptr, nullptr, vb, M, C, C, ACT_NONE, 0);
    attn_kernel<<<attnBlocks, 256, 0, stream>>>(qb, kb, vb, yb, T, H, Dh);
    gemm_wmma<<<gC, 256, 0, stream>>>(yb, WoT + (size_t)l * tCC, bo + (size_t)l * C,
                                      x, x, nullptr, M, C, C, ACT_NONE, 0);
    // reference reuses ln1 for the MLP branch
    ln_kernel<<<M, 256, 0, stream>>>(x, w1l, b1l, hbuf, C);
    gemm_wmma<<<gF, 256, 0, stream>>>(hbuf, W1T + (size_t)l * tCF, b1 + (size_t)l * F,
                                      nullptr, nullptr, hid, M, F, C, ACT_GELU, 0);
    gemm_wmma<<<gC, 256, 0, stream>>>(hid, W2T + (size_t)l * tCF, b2 + (size_t)l * C,
                                      x, x, nullptr, M, C, F, ACT_NONE, 0);
  }

  ln_kernel<<<M, 256, 0, stream>>>(x, lnfw, lnfb, hbuf, C);
  dim3 gH((V + 63) / 64, M / 64);
  gemm_wmma<<<gH, 256, 0, stream>>>(hbuf, hWT, nullptr, nullptr,
                                    logits, nullptr, M, V, C, ACT_NONE, 1);
  zero_loss<<<1, 1, 0, stream>>>(loss);
  loss_kernel<<<M, 256, 0, stream>>>(logits, targets, loss, V, 1.0f / (float)M);
}
